// URDS_43525198578114
// MI455X (gfx1250) — matrix operand order repaired and checked by reading, not compiled, not run on hardware
//
#include <hip/hip_runtime.h>
#include <hip/hip_bf16.h>
#include <math.h>

// ---------------- problem constants ----------------
#define CIN   128
#define COUT  128
#define HH    224
#define WW    224
#define HO    112
#define WO    112
#define PLANE (HH*WW)         // 50176
#define KTOT  1152            // 9 * CIN, ordered k = tap*128 + ci
#define KBLK  36              // KTOT / 32
#define PIX_PER_IMG (HO*WO)   // 12544
#define NIMG  8
#define NPIX  (NIMG*PIX_PER_IMG) // 100352
#define NTILE 64              // pixels per block
#define LDSB_STRIDE 40        // 32 K-slots + 8 pad (ushort) -> conflict-free b128 reads
#define BUFELEMS (NTILE*LDSB_STRIDE)

typedef __attribute__((ext_vector_type(16))) __bf16 v16bf;
typedef __attribute__((ext_vector_type(8)))  float  v8f;

union AFrag { uint4 q[2]; v16bf v; };

__device__ __forceinline__ unsigned short f2bf(float f) {
  unsigned int u = __float_as_uint(f);
  u += 0x7FFFu + ((u >> 16) & 1u);   // round-to-nearest-even
  return (unsigned short)(u >> 16);
}

// ---------------------------------------------------------------------------
// Kernel 0: pack conv weights f32 [co][ci][3][3] -> bf16 WMMA A-fragments.
// K order: k = tap*128 + ci  (tap = kh*3+kw)  => ci = k&127, tap = k>>7.
// wp[(kb*128 + co)*32 + slot]; slot order matches 16-bit A 16x32 VGPR layout:
// half0 -> K{0..7,16..23}, half1 -> K{8..15,24..31}.
// ---------------------------------------------------------------------------
__global__ void urds_pack_w(const float* __restrict__ w, unsigned short* __restrict__ wp) {
  int idx = blockIdx.x * blockDim.x + threadIdx.x;
  if (idx >= KBLK * COUT * 32) return;
  int slot = idx & 31;
  int co   = (idx >> 5) & 127;
  int kb   = idx >> 12;                  // / (128*32)
  int g = slot >> 4, t = slot & 15;
  int kin = (g == 0) ? ((t < 8) ? t : t + 8)
                     : ((t < 8) ? t + 8 : t + 16);
  int k   = kb * 32 + kin;
  int ci  = k & 127;
  int tap = k >> 7;                      // 0..8
  wp[idx] = f2bf(w[(co * CIN + ci) * 9 + tap]);
}

// ---------------------------------------------------------------------------
// Kernel 1: implicit-GEMM conv (stride 2, pad 1) + bias + exact GELU -> rq (ws)
// Block: 256 threads (8 waves). Block tile: 128 co x 64 pixels.
// Wave wv: co [wv*16, wv*16+16) x 64 pixels -> 4 WMMA accumulators.
// Double-buffered LDS B tile; one barrier per K-step; branchless border handling.
// ---------------------------------------------------------------------------
__global__ void __launch_bounds__(256)
urds_conv_gelu(const float* __restrict__ r, const unsigned short* __restrict__ wp,
               const float* __restrict__ bias, float* __restrict__ rq) {
  __shared__ unsigned short actB[2 * BUFELEMS];   // 10 KB

  int tid  = threadIdx.x;
  int lane = tid & 31;
  int wv   = tid >> 5;
  int half = lane >> 4;
  int l15  = lane & 15;

  int pixBase = blockIdx.x * NTILE;
  int coBase  = wv * 16;
  int aRow    = coBase + l15;

  // ---- per-thread staging state (loop-invariant pixel math) ----
  int pix  = tid & 63;
  int g    = tid >> 6;                       // 0..3 -> covers kk pairs
  int p    = pixBase + pix;
  int img  = p / PIX_PER_IMG;
  int prem = p - img * PIX_PER_IMG;
  int oh   = prem / WO;
  int ow   = prem - oh * WO;
  int ih0  = 2 * oh - 1;
  int iw0  = 2 * ow - 1;
  const float* rimg = r + (size_t)img * (CIN * PLANE);
  int sbase  = ih0 * WW + iw0;               // may be negative; masked per tap
  int ldsrow = pix * LDSB_STRIDE;

  // 9-bit tap-validity mask (border handling hoisted out of the K loop)
  unsigned okmask = 0;
#pragma unroll
  for (int tap = 0; tap < 9; ++tap) {
    int kh = (tap * 11) >> 5;                // tap/3
    int kw = tap - 3 * kh;
    bool ok = ((unsigned)(ih0 + kh) < (unsigned)HH) &&
              ((unsigned)(iw0 + kw) < (unsigned)WW);
    okmask |= (ok ? 1u : 0u) << tap;
  }

  v8f acc[4] = {};

  // ---- staging: 64 px x 32 K, channel-pairs packed into one b32 store ----
  auto stage = [&](int kb, unsigned short* buf) {
#pragma unroll
    for (int j = 0; j < 4; ++j) {
      int kk  = g * 2 + j * 8;               // even; pair (kk, kk+1)
      int k   = kb * 32 + kk;
      int tap = k >> 7;                      // 0..8 (pair shares tap)
      int ci  = k & 127;                     // even; pair = (ci, ci+1)
      int kh  = (tap * 11) >> 5;
      int kw  = tap - 3 * kh;
      bool ok = (okmask >> tap) & 1u;
      int addr = ci * PLANE + sbase + kh * WW + kw;
      addr = ok ? addr : 0;                  // clamp: branchless, always legal
      float v0 = rimg[addr];
      float v1 = rimg[addr + PLANE];         // ci+1 (ci even => ci+1 < 128)
      v0 = ok ? v0 : 0.f;
      v1 = ok ? v1 : 0.f;
      unsigned u = (unsigned)f2bf(v0) | ((unsigned)f2bf(v1) << 16);
      *(unsigned*)&buf[ldsrow + kk] = u;
    }
  };

  stage(0, actB);
  __syncthreads();

  for (int kb = 0; kb < KBLK; ++kb) {
    unsigned short* cur = actB + (kb & 1) * BUFELEMS;
    unsigned short* nxt = actB + ((kb + 1) & 1) * BUFELEMS;
    if (kb + 1 < KBLK) {
      stage(kb + 1, nxt);                    // global loads overlap WMMAs below
      __builtin_prefetch(wp + ((kb + 1) * COUT + aRow) * 32, 0, 0);
    }

    // A fragment (pre-packed bf16 weights): two contiguous b128 loads
    AFrag a;
    const uint4* ap =
        reinterpret_cast<const uint4*>(wp + ((kb * COUT + aRow) * 32 + half * 16));
    a.q[0] = ap[0]; a.q[1] = ap[1];

#pragma unroll
    for (int n = 0; n < 4; ++n) {
      AFrag b;
      const uint4* bp =
          reinterpret_cast<const uint4*>(&cur[(n * 16 + l15) * LDSB_STRIDE + half * 8]);
      b.q[0] = bp[0]; b.q[1] = bp[2];
      acc[n] = __builtin_amdgcn_wmma_f32_16x16x32_bf16(false, a.v, false, b.v,
                                                       (short)0, acc[n], false, false);
    }
    __syncthreads();                         // cur fully read; nxt fully staged
  }

  // ---- epilogue: bias + exact GELU, scatter to NCHW rq ----
#pragma unroll
  for (int n = 0; n < 4; ++n) {
    int pn   = pixBase + n * 16 + l15;
    int imgn = pn / PIX_PER_IMG;
    int prn  = pn - imgn * PIX_PER_IMG;
    float* outb = rq + (size_t)imgn * (COUT * PIX_PER_IMG) + prn;
#pragma unroll
    for (int v = 0; v < 8; ++v) {
      int co = coBase + v + 8 * half;        // C/D layout: M = v + 8*(lane>=16)
      float x = acc[n][v] + bias[co];
      float gl = 0.5f * x * (1.f + erff(x * 0.7071067811865475f));
      outb[co * PIX_PER_IMG] = gl;
    }
  }
}

// ---------------------------------------------------------------------------
// Kernel 2: details = r - up(r_); 3x3 max (-inf border); 2x2 avg; out = r_ + d
// Block: one (plane, 32x32 output tile). Separable max via LDS.
// ---------------------------------------------------------------------------
__global__ void __launch_bounds__(256)
urds_post(const float* __restrict__ r, const float* __restrict__ rq,
          float* __restrict__ out) {
  __shared__ float D[66 * 72];   // details, rows = input rows 2*oh0-1 .. +64
  __shared__ float CM[66 * 64];  // column-wise 3-max

  int plane = blockIdx.x >> 4;           // b*128 + c
  int tile  = blockIdx.x & 15;
  int oh0 = (tile >> 2) * 32;
  int ow0 = (tile & 3) * 32;

  const float* rp = r  + (size_t)plane * PLANE;
  const float* qp = rq + (size_t)plane * PIX_PER_IMG;
  float*       op = out + (size_t)plane * PIX_PER_IMG;

  int tid = threadIdx.x;

  for (int idx = tid; idx < 66 * 66; idx += 256) {
    int i = idx / 66, j = idx - i * 66;
    int ih = 2 * oh0 - 1 + i;
    int iw = 2 * ow0 - 1 + j;
    float d = -INFINITY;                              // geodesic border
    if ((unsigned)ih < (unsigned)HH && (unsigned)iw < (unsigned)WW)
      d = rp[ih * WW + iw] - qp[(ih >> 1) * WO + (iw >> 1)];
    D[i * 72 + j] = d;
  }
  __syncthreads();

  for (int idx = tid; idx < 66 * 64; idx += 256) {
    int i = idx >> 6, wc = idx & 63;
    float a = D[i * 72 + wc], b = D[i * 72 + wc + 1], c = D[i * 72 + wc + 2];
    CM[i * 64 + wc] = fmaxf(a, fmaxf(b, c));
  }
  __syncthreads();

  for (int idx = tid; idx < 1024; idx += 256) {
    int ohr = idx >> 5, owr = idx & 31;
    int oh = oh0 + ohr, ow = ow0 + owr;
    if (oh < HO && ow < WO) {
      float s = 0.f;
#pragma unroll
      for (int dh = 0; dh < 2; ++dh) {
        int hr = 2 * ohr + dh;
#pragma unroll
        for (int dw = 0; dw < 2; ++dw) {
          int wr = 2 * owr + dw;
          float m = fmaxf(CM[hr * 64 + wr],
                    fmaxf(CM[(hr + 1) * 64 + wr], CM[(hr + 2) * 64 + wr]));
          s += m;
        }
      }
      op[oh * WO + ow] = qp[oh * WO + ow] + 0.25f * s;
    }
  }
}

// ---------------------------------------------------------------------------
extern "C" void kernel_launch(void* const* d_in, const int* in_sizes, int n_in,
                              void* d_out, int out_size, void* d_ws, size_t ws_size,
                              hipStream_t stream) {
  (void)in_sizes; (void)n_in; (void)out_size; (void)ws_size;
  const float* r = (const float*)d_in[0];
  const float* w = (const float*)d_in[1];
  const float* b = (const float*)d_in[2];
  float* out = (float*)d_out;

  // workspace layout: [0, 288KB) packed bf16 weights; [512KB, +51.4MB) r_ (f32)
  unsigned short* wpack = (unsigned short*)d_ws;
  float* rq = (float*)((char*)d_ws + (512u << 10));

  int packN = KBLK * COUT * 32;                        // 147456
  urds_pack_w<<<packN / 256, 256, 0, stream>>>(w, wpack);
  urds_conv_gelu<<<NPIX / NTILE, 256, 0, stream>>>(r, wpack, b, rq); // 1568 blocks
  urds_post<<<NIMG * CIN * 16, 256, 0, stream>>>(r, rq, out);        // 16384 blocks
}
